// LabelPredictor_69801808495253
// MI455X (gfx1250) — matrix operand order repaired and checked by reading, not compiled, not run on hardware
//
#include <hip/hip_runtime.h>

// ---------------- problem constants (from reference) ----------------
#define NB     512   // batch n
#define SEQL   257   // L
#define HDIM   256   // h
#define LROWS  256   // l = L-1
#define K1     512   // 2*h
#define HIDN   256   // hidden
#define NLAB   50    // labels
#define NLABP  64    // padded labels
#define BM     64    // rows per block (stage-2 runs in two 32-row passes)
#define MTOT   (NB * LROWS)          // 131072
#define NBLK   (MTOT / BM)           // 2048
#define APAD   40    // LDS row stride for 32-K chunks (20 dwords -> conflict-free)
#define HSTR   264   // LDS row stride for hid (132 dwords = 4 mod 64 -> conflict-free)

typedef __attribute__((ext_vector_type(16))) __bf16 v16bf;
typedef __attribute__((ext_vector_type(8)))  float  v8f;

// ---- fp32 <-> bf16 split helpers (RNE) ----
__device__ __forceinline__ unsigned short f32_bf16(float f) {
    unsigned int u = __builtin_bit_cast(unsigned int, f);
    u += 0x7FFFu + ((u >> 16) & 1u);
    return (unsigned short)(u >> 16);
}
__device__ __forceinline__ float bf16_f32(unsigned short h) {
    return __builtin_bit_cast(float, (unsigned int)h << 16);
}
__device__ __forceinline__ __bf16 us2bf(unsigned short u) {
    return __builtin_bit_cast(__bf16, u);
}

// A-operand K permutation: fragment order {0-7,16-23 | 8-15,24-31} -> contiguous
__device__ __forceinline__ int permA(int k) {
    return (k & 7) | ((k >> 1) & 8) | ((k << 1) & 16);
}

// Load 16 contiguous bf16 (32B) as two aligned 16B chunks -> v16bf
struct bf16x16 { uint4 a, b; };
__device__ __forceinline__ v16bf load16(const __bf16* p) {
    bf16x16 t;
    t.a = *(const uint4*)p;
    t.b = *(const uint4*)(p + 8);
    return __builtin_bit_cast(v16bf, t);
}

__device__ __forceinline__ v8f wmma_bf16(v16bf a, v16bf b, v8f c) {
    return __builtin_amdgcn_wmma_f32_16x16x32_bf16(
        false, a, false, b, (short)0, c, false, false);
}

// ================= prep: split W1/W2 into bf16 hi/lo planes (ws) =================
__global__ __launch_bounds__(256)
void LabelPredictor_prep_kernel(const float* __restrict__ W1,
                                const float* __restrict__ W2,
                                __bf16* __restrict__ w1h, __bf16* __restrict__ w1l,
                                __bf16* __restrict__ w2h, __bf16* __restrict__ w2l)
{
    const int idx = blockIdx.x * 256 + threadIdx.x;
    if (idx < HIDN * K1) {                       // W1: (256,512) row-major, kept as-is
        const float v = W1[idx];
        const unsigned short h = f32_bf16(v);
        w1h[idx] = us2bf(h);
        w1l[idx] = us2bf(f32_bf16(v - bf16_f32(h)));
    } else {                                     // W2: (50,256) padded to (64,256)
        const int j = idx - HIDN * K1;           // [0, 64*256)
        const int n = j >> 8;
        const int k = j & 255;
        const float v = (n < NLAB) ? W2[n * HIDN + k] : 0.0f;
        const unsigned short h = f32_bf16(v);
        w2h[j] = us2bf(h);
        w2l[j] = us2bf(f32_bf16(v - bf16_f32(h)));
    }
}

// LDS: stage-1 staging unioned with stage-2 hid (32-row half)
struct __align__(16) SMem {
    union {
        struct {
            __bf16 cath[BM][APAD];      // gathered cat_f chunk, hi plane (k contiguous)
            __bf16 catl[BM][APAD];      // lo plane
            __bf16 w1h[HIDN][APAD];     // W1 chunk hi plane [n][k]
            __bf16 w1l[HIDN][APAD];     // W1 chunk lo plane
        } s1;                           // 51200 B
        struct {
            __bf16 hidh[32][HSTR];      // relu(hid) half, hi plane (k contiguous)
            __bf16 hidl[32][HSTR];      // lo plane
        } s2;                           // 33792 B
    };
};

__global__ __launch_bounds__(256)
void LabelPredictor_69801808495253_kernel(
        const float* __restrict__ feat,      // (512,257,256)
        const long long* __restrict__ heads, // (512,256) int64
        const __bf16* __restrict__ w1h, const __bf16* __restrict__ w1l,
        const float* __restrict__ b1,        // (256)
        const __bf16* __restrict__ w2h, const __bf16* __restrict__ w2l,
        const float* __restrict__ b2,        // (50)
        float* __restrict__ out)             // (512,256,50)
{
    __shared__ SMem smem;

    const int tid  = threadIdx.x;
    const int lane = tid & 31;
    const int wid  = tid >> 5;          // 8 waves
    const int l16  = lane & 15;
    const int hsel = (lane & 16);       // 0 | 16: operand element offset

    // stage-1 wave tiling: 2 row-halves (2 stripes each) x 4 col-groups of 64
    const int mp = wid & 1;             // row half: rows [32*mp, 32*mp+32)
    const int cg = wid >> 1;            // 64-col group

    v8f acc[2][4];
    #pragma unroll
    for (int s = 0; s < 2; ++s)
        #pragma unroll
        for (int t = 0; t < 4; ++t) acc[s][t] = v8f{};

    const int g0 = blockIdx.x * BM;

    // ================= Stage 1: hid = relu(cat_f @ W1^T + b1) =================
    for (int kc = 0; kc < K1 / 32; ++kc) {
        __syncthreads();   // prior compute done before overwriting staging

        // --- gathered cat_f chunk: split f32 -> bf16 hi/lo, k contiguous ---
        for (int idx = tid; idx < BM * 32; idx += 256) {
            const int m  = idx >> 5;
            const int kk = idx & 31;
            const int g  = g0 + m;
            const int i  = g >> 8;
            const int j  = g & 255;
            const int kg = kc * 32 + kk;
            float v;
            if (kg < HDIM) {                // dependent: feat[i, j+1, kg]
                v = feat[((size_t)i * SEQL + (j + 1)) * HDIM + kg];
            } else {                        // head: feat[i, heads[i,j], kg-256]
                const int hp = (int)heads[(size_t)i * LROWS + j];
                v = feat[((size_t)i * SEQL + hp) * HDIM + (kg - HDIM)];
            }
            const unsigned short h = f32_bf16(v);
            const int p = kk;
            smem.s1.cath[m][p] = us2bf(h);
            smem.s1.catl[m][p] = us2bf(f32_bf16(v - bf16_f32(h)));
        }

        // --- W1 chunk: pure b128 copy of pre-split planes (no conversion) ---
        {
            const int n = tid;              // one output neuron per thread
            const uint4* gh = (const uint4*)(w1h + (size_t)n * K1 + kc * 32);
            const uint4* gl = (const uint4*)(w1l + (size_t)n * K1 + kc * 32);
            uint4* lh = (uint4*)&smem.s1.w1h[n][0];
            uint4* ll = (uint4*)&smem.s1.w1l[n][0];
            #pragma unroll
            for (int q = 0; q < 4; ++q) { lh[q] = gh[q]; ll[q] = gl[q]; }
            if (kc + 1 < K1 / 32)
                __builtin_prefetch(w1h + (size_t)n * K1 + (kc + 1) * 32, 0, 0);
        }

        __syncthreads();

        // --- A fragments for both 16-row stripes of this wave's half ---
        const v16bf ah0 = load16(&smem.s1.cath[32 * mp + l16][hsel]);
        const v16bf al0 = load16(&smem.s1.catl[32 * mp + l16][hsel]);
        const v16bf ah1 = load16(&smem.s1.cath[32 * mp + 16 + l16][hsel]);
        const v16bf al1 = load16(&smem.s1.catl[32 * mp + 16 + l16][hsel]);

        // --- 4 col tiles; each B fragment pair feeds 6 WMMAs ---
        #pragma unroll
        for (int t = 0; t < 4; ++t) {
            const int n = 64 * cg + 16 * t + l16;
            const v16bf bh = load16(&smem.s1.w1h[n][hsel]);
            const v16bf bl = load16(&smem.s1.w1l[n][hsel]);
            acc[0][t] = wmma_bf16(ah0, bh, acc[0][t]);
            acc[0][t] = wmma_bf16(al0, bh, acc[0][t]);
            acc[0][t] = wmma_bf16(ah0, bl, acc[0][t]);
            acc[1][t] = wmma_bf16(ah1, bh, acc[1][t]);
            acc[1][t] = wmma_bf16(al1, bh, acc[1][t]);
            acc[1][t] = wmma_bf16(ah1, bl, acc[1][t]);
        }
    }

    // ====== Stage 2 in two 32-row passes, reusing the 32-row hid buffer ======
    const int mr2 = wid & 1;
    const int nc2 = wid >> 1;

    for (int half = 0; half < 2; ++half) {
        __syncthreads();   // staging / previous-pass reads done; hid may be written

        // --- bias + relu, split-pack this half's hid rows (k contiguous) ---
        if (mp == half) {
            #pragma unroll
            for (int s = 0; s < 2; ++s) {
                #pragma unroll
                for (int t = 0; t < 4; ++t) {
                    const int n  = 64 * cg + 16 * t + l16;
                    const float bb = b1[n];
                    const int cs = n;
                    #pragma unroll
                    for (int v = 0; v < 8; ++v) {
                        const int row = 16 * s + v + (hsel >> 1);  // local 0..31
                        const float hv = fmaxf(acc[s][t][v] + bb, 0.0f);
                        const unsigned short h = f32_bf16(hv);
                        smem.s2.hidh[row][cs] = us2bf(h);
                        smem.s2.hidl[row][cs] = us2bf(f32_bf16(hv - bf16_f32(h)));
                    }
                }
            }
        }
        __syncthreads();

        // --- out = hid @ W2^T + b2 (N padded to 64, rows 50..63 pre-zeroed) ---
        v8f acc2 = v8f{};
        #pragma unroll
        for (int kc = 0; kc < HIDN / 32; ++kc) {
            const v16bf ah = load16(&smem.s2.hidh[16 * mr2 + l16][32 * kc + hsel]);
            const v16bf al = load16(&smem.s2.hidl[16 * mr2 + l16][32 * kc + hsel]);
            const int n = 16 * nc2 + l16;
            const v16bf bh = load16(&w2h[(size_t)n * HIDN + 32 * kc + hsel]);
            const v16bf bl = load16(&w2l[(size_t)n * HIDN + 32 * kc + hsel]);
            acc2 = wmma_bf16(ah, bh, acc2);
            acc2 = wmma_bf16(al, bh, acc2);
            acc2 = wmma_bf16(ah, bl, acc2);
        }

        // --- epilogue: bias + masked store of 50 label columns ---
        const int n = 16 * nc2 + l16;
        if (n < NLAB) {
            const float bb = b2[n];
            #pragma unroll
            for (int v = 0; v < 8; ++v) {
                const int row = 16 * mr2 + v + (hsel >> 1);       // local 0..31
                const int g   = g0 + 32 * half + row;
                out[(size_t)g * NLAB + n] = acc2[v] + bb;
            }
        }
    }
}

extern "C" void kernel_launch(void* const* d_in, const int* in_sizes, int n_in,
                              void* d_out, int out_size, void* d_ws, size_t ws_size,
                              hipStream_t stream) {
    (void)in_sizes; (void)n_in; (void)out_size; (void)ws_size;
    const float*     feat  = (const float*)d_in[0];
    const long long* heads = (const long long*)d_in[1];
    // d_in[2] = masks (all ones; unused by the reference output)
    const float*     W1    = (const float*)d_in[3];
    const float*     b1    = (const float*)d_in[4];
    const float*     W2    = (const float*)d_in[5];
    const float*     b2    = (const float*)d_in[6];
    float*           out   = (float*)d_out;

    // workspace layout: pre-split bf16 hi/lo weight planes (~576 KB)
    __bf16* w1h = (__bf16*)d_ws;
    __bf16* w1l = w1h + (size_t)HIDN * K1;
    __bf16* w2h = w1l + (size_t)HIDN * K1;
    __bf16* w2l = w2h + (size_t)NLABP * HIDN;

    const int prep_elems = HIDN * K1 + NLABP * HIDN;   // 147456
    LabelPredictor_prep_kernel<<<prep_elems / 256, 256, 0, stream>>>(
        W1, W2, w1h, w1l, w2h, w2l);

    LabelPredictor_69801808495253_kernel<<<NBLK, 256, 0, stream>>>(
        feat, heads, w1h, w1l, b1, w2h, w2l, b2, out);
}
